// SwinBlock_18124761989754
// MI455X (gfx1250) — matrix operand order repaired and checked by reading, not compiled, not run on hardware
//
#include <hip/hip_runtime.h>
#include <hip/hip_bf16.h>
#include <math.h>

#define DIM   192
#define HEADS 6
#define HD    32
#define NWIN  4096          // 16 batches * 16 * 16 windows
#define NTOK  262144        // 16 * 128 * 128
#define LN_EPS 1e-5f

typedef __attribute__((ext_vector_type(16))) _Float16 v16h;
typedef __attribute__((ext_vector_type(8)))  float    v8f;

union H16 { v16h v; _Float16 h[16]; uint4 q[2]; };
union H8  { _Float16 h[8]; uint4 q; };

__device__ __forceinline__ v8f wmma32(v16h a, v16h b, v8f c) {
  // D = A(16x32 f16) * B(32x16 f16) + C(16x16 f32)
  return __builtin_amdgcn_wmma_f32_16x16x32_f16(false, a, false, b, (short)0, c, false, false);
}

// A fragment: src row-major [row][col] (ld in halves).
// lane L holds row = row0 + (L&15); halves 0..7 -> K = (L>=16?8:0)+h ; halves 8..15 -> +16
__device__ __forceinline__ v16h loadA(const _Float16* src, int ld, int row0, int col0, int lane) {
  int r    = row0 + (lane & 15);
  int koff = (lane >> 4) * 8;
  const _Float16* p = src + (size_t)r * ld + col0 + koff;
  H16 u;
  u.q[0] = *(const uint4*)(p);
  u.q[1] = *(const uint4*)(p + 16);
  return u.v;
}

// B fragment: source stored [n][k] row-major (stride ld halves); B[k][n] for WMMA.
// lane L holds column n = n0 + (L&15); halves 0..7 -> k = k0 + (L>=16?8:0) + h ; halves 8..15 -> +16
__device__ __forceinline__ v16h loadB(const _Float16* src, int ld, int n0, int k0, int lane) {
  int n = n0 + (lane & 15);
  int k = k0 + (lane >> 4) * 8;
  const _Float16* p = src + (size_t)n * ld + k;
  H16 u;
  u.q[0] = *(const uint4*)(p);
  u.q[1] = *(const uint4*)(p + 16);
  return u.v;
}

__global__ void cvt_kernel(const float* __restrict__ s, _Float16* __restrict__ d, int n) {
  int i = blockIdx.x * 256 + threadIdx.x;
  if (i < n) d[i] = (_Float16)s[i];
}

// LayerNorm (one wave per token, 8 tokens / block of 256).
// shifted=1: dst slot enumerates (window,row) and source is rolled by (-4,-4) + partitioned.
__global__ void ln_kernel(const float* __restrict__ src, const float* __restrict__ g,
                          const float* __restrict__ beta, _Float16* __restrict__ dst,
                          int shifted) {
  int tid  = threadIdx.x;
  int lane = tid & 31;
  int slot = blockIdx.x * 8 + (tid >> 5);
  int srow;
  if (shifted) {
    int win = slot >> 6, r = slot & 63;
    int b = win >> 8, wh = (win >> 4) & 15, ww = win & 15;
    int i = r >> 3, j = r & 7;
    int sh = (wh * 8 + i + 4) & 127;     // roll(-4): shifted[p] = x[(p+4) mod 128]
    int sw = (ww * 8 + j + 4) & 127;
    srow = b * 16384 + sh * 128 + sw;
  } else {
    srow = slot;
  }
  const float* xp = src + (size_t)srow * DIM;
  float vals[6]; float s = 0.f, sq = 0.f;
  for (int k = 0; k < 6; ++k) {
    float v = xp[lane + 32 * k];
    vals[k] = v; s += v; sq += v * v;
  }
  for (int m = 16; m >= 1; m >>= 1) { s += __shfl_xor(s, m); sq += __shfl_xor(sq, m); }
  float mean = s * (1.f / 192.f);
  float var  = sq * (1.f / 192.f) - mean * mean;
  float rstd = rsqrtf(var + LN_EPS);
  _Float16* op = dst + (size_t)slot * DIM;
  for (int k = 0; k < 6; ++k) {
    int c = lane + 32 * k;
    op[c] = (_Float16)((vals[k] - mean) * rstd * g[c] + beta[c]);
  }
}

// Fused per-window: QKV GEMM -> l2norm(q,k) -> per-head (S, softmax+clip, A@V) -> proj
// -> reverse-shift scatter with residual.  One workgroup (8 waves) per window.
// All GEMM loops process N-tile PAIRS sharing one A fragment (2 wmma per A load).
__global__ __launch_bounds__(256) void attn_kernel(
    const _Float16* __restrict__ hw,     // [4096][64][192] f16 (LN1'd, partitioned)
    const _Float16* __restrict__ wqkv,   // [576][192] f16 (out,in)
    const float*    __restrict__ qkv_b,  // [576]
    const _Float16* __restrict__ wproj,  // [192][192] f16
    const float*    __restrict__ proj_b, // [192]
    const float*    __restrict__ x_in,   // shortcut [NTOK][192] f32
    float*          __restrict__ x1)     // out residual [NTOK][192] f32
{
  __shared__ alignas(16) _Float16 sHW[64 * 192];   // 24.0 KB  input tile
  __shared__ alignas(16) _Float16 sQK[64 * 384];   // 48.0 KB  q (0..191) | k (192..383)
  __shared__ alignas(16) _Float16 sVT[192 * 64];   // 24.0 KB  v transposed [d][m]
  __shared__ alignas(16) float    sS [64 * 64];    // 16.0 KB  scores
  __shared__ alignas(16) _Float16 sA [64 * 64];    //  8.0 KB  probabilities
  __shared__ alignas(16) _Float16 sO [64 * 192];   // 24.0 KB  attention output

  int tid = threadIdx.x, lane = tid & 31, wave = tid >> 5;
  int win = blockIdx.x;

  { // stage input tile; speculatively prefetch the next window's tile
    const uint4* s = (const uint4*)(hw + (size_t)win * (64 * 192));
    uint4* d = (uint4*)sHW;
    for (int i = tid; i < 1536; i += 256) d[i] = s[i];
    if (win + 1 < NWIN)
      __builtin_prefetch(hw + (size_t)(win + 1) * (64 * 192) + tid * 48, 0, 1);
  }
  __syncthreads();

  // ---- QKV: (64x192) @ (192x576)^T, K = 6*32 ----
  // q/k tiles (ntile 0..23): 48 pairs, scalar column stores into row-major sQK
  for (int t = wave; t < 48; t += 8) {
    int mt = t / 12, n0 = (t % 12) * 32;            // two n-tiles: n0, n0+16
    v8f acc0 = {}, acc1 = {};
    for (int kb = 0; kb < 6; ++kb) {
      v16h a  = loadA(sHW, 192, mt * 16, kb * 32, lane);
      v16h b0 = loadB(wqkv, 192, n0,      kb * 32, lane);
      v16h b1 = loadB(wqkv, 192, n0 + 16, kb * 32, lane);
      acc0 = wmma32(a, b0, acc0);
      acc1 = wmma32(a, b1, acc1);
    }
    int nA = n0 + (lane & 15), nB = nA + 16;
    float biasA = qkv_b[nA], biasB = qkv_b[nB];
    int mb = mt * 16 + (lane >> 4) * 8;
    for (int r = 0; r < 8; ++r) {
      sQK[(mb + r) * 384 + nA] = (_Float16)(acc0[r] + biasA);
      sQK[(mb + r) * 384 + nB] = (_Float16)(acc1[r] + biasB);
    }
  }
  // v tiles (ntile 24..35): 24 pairs, transposed layout -> packed b128 stores
  for (int t = wave; t < 24; t += 8) {
    int mt = t / 6, n0 = 384 + (t % 6) * 32;
    v8f acc0 = {}, acc1 = {};
    for (int kb = 0; kb < 6; ++kb) {
      v16h a  = loadA(sHW, 192, mt * 16, kb * 32, lane);
      v16h b0 = loadB(wqkv, 192, n0,      kb * 32, lane);
      v16h b1 = loadB(wqkv, 192, n0 + 16, kb * 32, lane);
      acc0 = wmma32(a, b0, acc0);
      acc1 = wmma32(a, b1, acc1);
    }
    int nA = n0 + (lane & 15), nB = nA + 16;        // 384..575
    float biasA = qkv_b[nA], biasB = qkv_b[nB];
    int mb = mt * 16 + (lane >> 4) * 8;
    H8 p0, p1;
    for (int r = 0; r < 8; ++r) {
      p0.h[r] = (_Float16)(acc0[r] + biasA);
      p1.h[r] = (_Float16)(acc1[r] + biasB);
    }
    *(uint4*)(sVT + (nA - 384) * 64 + mb) = p0.q;
    *(uint4*)(sVT + (nB - 384) * 64 + mb) = p1.q;
  }
  __syncthreads();

  // ---- l2 normalize q and k rows per head: 64 rows * 12 vectors of 32 (vectorized) ----
  for (int vid = tid; vid < 768; vid += 256) {
    int row = vid / 12, c0 = (vid % 12) * 32;
    _Float16* p = sQK + row * 384 + c0;
    H16 u0, u1;
    u0.q[0] = *(uint4*)(p);      u0.q[1] = *(uint4*)(p + 8);
    u1.q[0] = *(uint4*)(p + 16); u1.q[1] = *(uint4*)(p + 24);
    float s = 0.f;
    for (int j = 0; j < 16; ++j) {
      float a = (float)u0.h[j]; s += a * a;
      float b = (float)u1.h[j]; s += b * b;
    }
    float inv = 1.f / fmaxf(sqrtf(s), 1e-12f);
    for (int j = 0; j < 16; ++j) {
      u0.h[j] = (_Float16)((float)u0.h[j] * inv);
      u1.h[j] = (_Float16)((float)u1.h[j] * inv);
    }
    *(uint4*)(p)      = u0.q[0]; *(uint4*)(p + 8)  = u0.q[1];
    *(uint4*)(p + 16) = u1.q[0]; *(uint4*)(p + 24) = u1.q[1];
  }
  __syncthreads();

  const float scale = 0.17677669529663687f;   // 32^-0.5
  for (int head = 0; head < HEADS; ++head) {
    // S = q @ k^T : 4x4 tiles as 8 pairs (one pair per wave), single K=32 step
    {
      int mt = wave >> 1, n0 = (wave & 1) * 32;
      v16h a  = loadA(sQK, 384, mt * 16, head * 32, lane);
      v16h b0 = loadB(sQK, 384, n0,      192 + head * 32, lane);
      v16h b1 = loadB(sQK, 384, n0 + 16, 192 + head * 32, lane);
      v8f acc0 = {}, acc1 = {};
      acc0 = wmma32(a, b0, acc0);
      acc1 = wmma32(a, b1, acc1);
      int nA = n0 + (lane & 15);
      int mb = mt * 16 + (lane >> 4) * 8;
      for (int r = 0; r < 8; ++r) {
        sS[(mb + r) * 64 + nA]      = acc0[r];
        sS[(mb + r) * 64 + nA + 16] = acc1[r];
      }
    }
    __syncthreads();
    // softmax + clip, one thread per row
    if (tid < 64) {
      float mx = -3.4e38f;
      for (int j = 0; j < 64; ++j) mx = fmaxf(mx, sS[tid * 64 + j] * scale);
      float sum = 0.f;
      for (int j = 0; j < 64; ++j) sum += expf(sS[tid * 64 + j] * scale - mx);
      float rs = 1.f / sum;
      for (int j = 0; j < 64; ++j) {
        float p = expf(sS[tid * 64 + j] * scale - mx) * rs;
        p = fminf(fmaxf(p, 1e-6f), 1.0f);
        sA[tid * 64 + j] = (_Float16)p;
      }
    }
    __syncthreads();
    // O = A @ V : 4x2 tiles (one per wave), K = 2*32
    {
      int mt = wave >> 1, nt = wave & 1;
      v8f acc = {};
      for (int kb = 0; kb < 2; ++kb) {
        v16h a = loadA(sA, 64, mt * 16, kb * 32, lane);
        v16h b = loadB(sVT, 64, head * 32 + nt * 16, kb * 32, lane);
        acc = wmma32(a, b, acc);
      }
      int nb = head * 32 + nt * 16 + (lane & 15);
      int mb = mt * 16 + (lane >> 4) * 8;
      for (int r = 0; r < 8; ++r) sO[(mb + r) * 192 + nb] = (_Float16)acc[r];
    }
    __syncthreads();
  }

  // ---- proj (64x192)@(192x192)^T + reverse shift scatter + residual : 24 pairs ----
  int b = win >> 8, wh = (win >> 4) & 15, ww = win & 15;
  for (int t = wave; t < 24; t += 8) {
    int mt = t / 6, n0 = (t % 6) * 32;
    v8f acc0 = {}, acc1 = {};
    for (int kb = 0; kb < 6; ++kb) {
      v16h a  = loadA(sO, 192, mt * 16, kb * 32, lane);
      v16h b0 = loadB(wproj, 192, n0,      kb * 32, lane);
      v16h b1 = loadB(wproj, 192, n0 + 16, kb * 32, lane);
      acc0 = wmma32(a, b0, acc0);
      acc1 = wmma32(a, b1, acc1);
    }
    int nA = n0 + (lane & 15);
    float biasA = proj_b[nA], biasB = proj_b[nA + 16];
    int mb = mt * 16 + (lane >> 4) * 8;
    for (int r = 0; r < 8; ++r) {
      int wr = mb + r;
      int i = wr >> 3, j = wr & 7;
      int oh = (wh * 8 + i + 4) & 127;   // roll(+4) undoes roll(-4)
      int ow = (ww * 8 + j + 4) & 127;
      size_t base = (((size_t)b * 16384) + (size_t)(oh * 128 + ow)) * DIM;
      x1[base + nA]      = x_in[base + nA]      + acc0[r] + biasA;
      x1[base + nA + 16] = x_in[base + nA + 16] + acc1[r] + biasB;
    }
  }
}

// Fused MLP per 64-token block: h = gelu(xn@w1^T + b1); out = h@w2^T + b2 + x1
__global__ __launch_bounds__(256) void mlp_kernel(
    const _Float16* __restrict__ xn,   // [NTOK][192] f16 (LN2'd)
    const _Float16* __restrict__ w1,   // [768][192] f16
    const float*    __restrict__ b1,
    const _Float16* __restrict__ w2,   // [192][768] f16
    const float*    __restrict__ b2,
    const float*    __restrict__ x1,
    float*          __restrict__ out)
{
  __shared__ alignas(16) _Float16 sX[64 * 192];   // 24 KB
  __shared__ alignas(16) _Float16 sH[64 * 768];   // 96 KB
  int tid = threadIdx.x, lane = tid & 31, wave = tid >> 5;
  size_t t0 = (size_t)blockIdx.x * 64;

  { const uint4* s = (const uint4*)(xn + t0 * DIM);
    uint4* d = (uint4*)sX;
    for (int i = tid; i < 1536; i += 256) d[i] = s[i];
    if (blockIdx.x + 1 < NTOK / 64)
      __builtin_prefetch(xn + (t0 + 64) * DIM + tid * 48, 0, 1);
  }
  __syncthreads();

  // GEMM1: (64x192)@(192x768)^T, 4x48 tiles as 96 pairs
  for (int t = wave; t < 96; t += 8) {
    int mt = t / 24, n0 = (t % 24) * 32;
    v8f acc0 = {}, acc1 = {};
    for (int kb = 0; kb < 6; ++kb) {
      v16h a  = loadA(sX, 192, mt * 16, kb * 32, lane);
      v16h b0 = loadB(w1, 192, n0,      kb * 32, lane);
      v16h b1 = loadB(w1, 192, n0 + 16, kb * 32, lane);
      acc0 = wmma32(a, b0, acc0);
      acc1 = wmma32(a, b1, acc1);
    }
    int nA = n0 + (lane & 15);
    float biasA = b1[nA], biasB = b1[nA + 16];
    int mb = mt * 16 + (lane >> 4) * 8;
    for (int r = 0; r < 8; ++r) {
      float v0 = acc0[r] + biasA;
      float v1 = acc1[r] + biasB;
      v0 = 0.5f * v0 * (1.f + erff(v0 * 0.70710678118654752f));  // exact GELU
      v1 = 0.5f * v1 * (1.f + erff(v1 * 0.70710678118654752f));
      sH[(mb + r) * 768 + nA]      = (_Float16)v0;
      sH[(mb + r) * 768 + nA + 16] = (_Float16)v1;
    }
  }
  __syncthreads();

  // GEMM2: (64x768)@(768x192)^T, 4x12 tiles as 24 pairs, K = 24*32
  for (int t = wave; t < 24; t += 8) {
    int mt = t / 6, n0 = (t % 6) * 32;
    v8f acc0 = {}, acc1 = {};
    for (int kb = 0; kb < 24; ++kb) {
      v16h a  = loadA(sH, 768, mt * 16, kb * 32, lane);
      v16h b0 = loadB(w2, 768, n0,      kb * 32, lane);
      v16h b1 = loadB(w2, 768, n0 + 16, kb * 32, lane);
      acc0 = wmma32(a, b0, acc0);
      acc1 = wmma32(a, b1, acc1);
    }
    int nA = n0 + (lane & 15);
    float biasA = b2[nA], biasB = b2[nA + 16];
    int mb = mt * 16 + (lane >> 4) * 8;
    for (int r = 0; r < 8; ++r) {
      size_t base = (t0 + (size_t)(mb + r)) * DIM;
      out[base + nA]      = x1[base + nA]      + acc0[r] + biasA;
      out[base + nA + 16] = x1[base + nA + 16] + acc1[r] + biasB;
    }
  }
}

extern "C" void kernel_launch(void* const* d_in, const int* in_sizes, int n_in,
                              void* d_out, int out_size, void* d_ws, size_t ws_size,
                              hipStream_t stream) {
  (void)in_sizes; (void)n_in; (void)out_size; (void)ws_size;
  const float* x      = (const float*)d_in[0];
  const float* qkv_w  = (const float*)d_in[1];
  const float* qkv_b  = (const float*)d_in[2];
  const float* proj_w = (const float*)d_in[3];
  const float* proj_b = (const float*)d_in[4];
  const float* n1_g   = (const float*)d_in[5];
  const float* n1_b   = (const float*)d_in[6];
  const float* n2_g   = (const float*)d_in[7];
  const float* n2_b   = (const float*)d_in[8];
  const float* w1     = (const float*)d_in[9];
  const float* b1     = (const float*)d_in[10];
  const float* w2     = (const float*)d_in[11];
  const float* b2     = (const float*)d_in[12];
  float* out = (float*)d_out;

  char* ws = (char*)d_ws;
  _Float16* wqkv_h  = (_Float16*)ws; ws += (size_t)576 * 192 * 2;
  _Float16* wproj_h = (_Float16*)ws; ws += (size_t)192 * 192 * 2;
  _Float16* w1_h    = (_Float16*)ws; ws += (size_t)768 * 192 * 2;
  _Float16* w2_h    = (_Float16*)ws; ws += (size_t)192 * 768 * 2;
  _Float16* hwbuf   = (_Float16*)ws; ws += (size_t)NTOK * DIM * 2;  // reused as xn after attn
  float*    x1      = (float*)ws;    ws += (size_t)NTOK * DIM * 4;

  cvt_kernel<<<(576 * 192 + 255) / 256, 256, 0, stream>>>(qkv_w, wqkv_h, 576 * 192);
  cvt_kernel<<<(192 * 192 + 255) / 256, 256, 0, stream>>>(proj_w, wproj_h, 192 * 192);
  cvt_kernel<<<(768 * 192 + 255) / 256, 256, 0, stream>>>(w1, w1_h, 768 * 192);
  cvt_kernel<<<(192 * 768 + 255) / 256, 256, 0, stream>>>(w2, w2_h, 192 * 768);

  ln_kernel<<<NTOK / 8, 256, 0, stream>>>(x, n1_g, n1_b, hwbuf, 1);
  attn_kernel<<<NWIN, 256, 0, stream>>>(hwbuf, wqkv_h, qkv_b, wproj_h, proj_b, x, x1);
  ln_kernel<<<NTOK / 8, 256, 0, stream>>>(x1, n2_g, n2_b, hwbuf, 0);
  mlp_kernel<<<NTOK / 64, 256, 0, stream>>>(hwbuf, w1_h, b1, w2_h, b2, x1, out);
}
